// PNALayer_68393059221827
// MI455X (gfx1250) — matrix operand order repaired and checked by reading, not compiled, not run on hardware
//
#include <hip/hip_runtime.h>
#include <hip/hip_bf16.h>
#include <math.h>

// ---------------------------------------------------------------------------
// PNA layer + GRU for MI455X (gfx1250, wave32, WMMA).
// Dominant cost: streaming adj_features (134 MB) once. Fused per-(b,i) WMMA
// kernel with double-buffered ASYNC global->LDS tile staging (ASYNCcnt).
// All GEMMs use v_wmma_f32_16x16x32_f16 with f32 accumulate.
// ---------------------------------------------------------------------------

typedef __attribute__((ext_vector_type(16))) _Float16 v16h;
typedef __attribute__((ext_vector_type(8)))  _Float16 v8h;
typedef __attribute__((ext_vector_type(8)))  float    v8f;
typedef __attribute__((ext_vector_type(4)))  int      v4i;

#define EPSV  1e-5f
#define LEAK  0.01f
#define AVGD  3.3f
#define BIGV  1e30f

// ---- async global->LDS copy (CDNA5 ASYNCcnt path, probed via __has_builtin)
#if defined(__gfx1250__) && __has_builtin(__builtin_amdgcn_global_load_async_to_lds_b128)
#define HAVE_ASYNC_LDS 1
#else
#define HAVE_ASYNC_LDS 0
#endif

#if HAVE_ASYNC_LDS
// builtin signature (from clang diagnostic): (v4i AS1*, v4i AS3*, Ii, Ii)
typedef __attribute__((address_space(1))) v4i as1_v4i;
typedef __attribute__((address_space(3))) v4i as3_v4i;
static __device__ __forceinline__ void async_cp16(float* l, const float* g) {
  __builtin_amdgcn_global_load_async_to_lds_b128((as1_v4i*)(void*)g,
                                                 (as3_v4i*)(void*)l, 0, 0);
}
#if __has_builtin(__builtin_amdgcn_s_wait_asynccnt)
#define WAIT_ASYNC(n) __builtin_amdgcn_s_wait_asynccnt(n)
#else
#define WAIT_ASYNC(n) asm volatile("s_wait_asynccnt " #n)
#endif
#else
static __device__ __forceinline__ void async_cp16(float* l, const float* g) {
  *(float4*)l = *(const float4*)g;   // sync fallback (barrier provides order)
}
#define WAIT_ASYNC(n) ((void)0)
#endif

// ---- WMMA fragment helpers (wave32) ---------------------------------------
// A (16x32 f16): lane L holds row M=L%16; lanes<16: K=0..15,
// lanes>=16: K=16..31 (contiguous-K per lane-half, matching B).
static __device__ __forceinline__ v16h load_fragA_f32(const float* p) {
  v16h r;
#pragma unroll
  for (int c = 0; c < 8; ++c) r[c] = (_Float16)p[c];
#pragma unroll
  for (int c = 0; c < 8; ++c) r[8 + c] = (_Float16)p[8 + c];
  return r;
}
// B (32x16 f16): lane L holds col N=L%16; lanes<16: K=0..15,
// lanes>=16: K=16..31 (contiguous-K per lane-half).
// p must already point at row(n) + kbase + ((lane>>4)<<4).
static __device__ __forceinline__ v16h load_fragB_f16(const _Float16* p) {
  v8h lo = *(const v8h*)p;
  v8h hi = *(const v8h*)(p + 8);
  return __builtin_shufflevector(lo, hi, 0,1,2,3,4,5,6,7,8,9,10,11,12,13,14,15);
}
static __device__ __forceinline__ v8f wmma_f16(v16h a, v16h b, v8f c) {
  return __builtin_amdgcn_wmma_f32_16x16x32_f16(false, a, false, b, (short)0, c,
                                                false, false);
}

// ---- weight prep: f32 (RxC) -> f16 transposed (CxR) -----------------------
__global__ void prep_transpose_f16(_Float16* __restrict__ dst,
                                   const float* __restrict__ src, int R, int C) {
  int idx = blockIdx.x * 256 + threadIdx.x;
  if (idx < R * C) {
    int r = idx / C, c = idx % C;
    dst[(size_t)c * R + r] = (_Float16)src[(size_t)r * C + c];
  }
}
__global__ void prep_cast_f16(_Float16* __restrict__ dst,
                              const float* __restrict__ src, int n) {
  int idx = blockIdx.x * 256 + threadIdx.x;
  if (idx < n) dst[idx] = (_Float16)src[idx];
}

// ---- generic GEMM: C[m,n] = act( sum_k A[m,k]*W[n,k] + bias[n] ) ----------
// A: f32 MxK row-major, W: f16 NxK row-major (transposed weight), K % 32 == 0.
// 4 waves / block, one 16x16 output tile per wave.
__global__ __launch_bounds__(128) void gemm_wmma(
    const float* __restrict__ A, const _Float16* __restrict__ W,
    const float* __restrict__ bias, float* __restrict__ C,
    int M, int N, int K, int act) {
  const int lane = threadIdx.x & 31;
  const int wid  = threadIdx.x >> 5;
  const int tile = blockIdx.x * 4 + wid;
  const int ntn  = N >> 4;
  const int mt = tile / ntn, nt = tile % ntn;
  if (mt >= (M >> 4)) return;
  const int m0 = mt << 4, n0 = nt << 4;
  const int half8  = (lane >> 4) << 3;
  const int half16 = (lane >> 4) << 4;

  const float*    ap = A + (size_t)(m0 + (lane & 15)) * K + half16;
  const _Float16* wp = W + (size_t)(n0 + (lane & 15)) * K + half16;

  v8f acc = {};
  for (int k0 = 0; k0 < K; k0 += 32) {
    v16h a = load_fragA_f32(ap + k0);
    v16h b = load_fragB_f16(wp + k0);
    acc = wmma_f16(a, b, acc);
  }
  const int col = n0 + (lane & 15);
  const float bv = bias ? bias[col] : 0.f;
#pragma unroll
  for (int v = 0; v < 8; ++v) {
    float y = acc[v] + bv;
    if (act == 1) y = y > 0.f ? y : LEAK * y;
    C[(size_t)(m0 + v + half8) * N + col] = y;
  }
}

// ---- fused PNA kernel: one block per (b,i) row ----------------------------
// E(256x128) = adj_features[b,i] (256x64) @ We (64x128) via WMMA, fused with
// masked mean/var/max/min over j. adj_features + hj tiles are double-buffered
// in LDS via async global->LDS copies. Writes the 1664-wide m_cat row into X.
__global__ __launch_bounds__(128) void pna_kernel(
    const float* __restrict__ input, const float* __restrict__ adj,
    const float* __restrict__ af,    const float* __restrict__ hi,
    const float* __restrict__ hj,    const float* __restrict__ pre_b,
    const _Float16* __restrict__ WeT, float* __restrict__ X) {
  __shared__ float saf[2][16 * 64];    // adj_features tile, 4KB per buffer
  __shared__ float shj[2][16 * 128];   // hj tile, 8KB per buffer
  __shared__ float sadj[256];
  __shared__ float sred[128];

  const int tid  = threadIdx.x;
  const int lane = tid & 31;
  const int w    = tid >> 5;
  const int row  = blockIdx.x;                 // b*256 + i
  const int b    = row >> 8;
  const float* afb = af + (size_t)row * (256 * 64);
  const float* hjp = hj + (size_t)b * 256 * 128;

  // stage adj row, deterministic tree-reduce degree
  float a0 = adj[(size_t)row * 256 + tid];
  float a1 = adj[(size_t)row * 256 + 128 + tid];
  sadj[tid] = a0; sadj[tid + 128] = a1;
  sred[tid] = a0 + a1;
  __syncthreads();
  for (int s = 64; s > 0; s >>= 1) {
    if (tid < s) sred[tid] += sred[tid + s];
    __syncthreads();
  }
  const float deg = sred[0];

  // per-lane column constants for this wave's two 16-wide n-tiles
  const int half8 = (lane >> 4) << 3;
  int   kcol[2];
  float cst[2];
  v16h  bf[2][2];
#pragma unroll
  for (int t = 0; t < 2; ++t) {
    kcol[t] = (w << 5) + (t << 4) + (lane & 15);
    cst[t]  = hi[(size_t)row * 128 + kcol[t]] + pre_b[kcol[t]];
#pragma unroll
    for (int kk = 0; kk < 2; ++kk)
      bf[t][kk] = load_fragB_f16(WeT + (size_t)kcol[t] * 64 + kk * 32 +
                                 ((lane >> 4) << 4));
  }

  float s1[2] = {0.f, 0.f}, s2[2] = {0.f, 0.f};
  float mx[2] = {-BIGV, -BIGV}, mn[2] = {BIGV, BIGV};

  // Both tiles are CONTIGUOUS blocks in global memory:
  //   af tile  = 16*64  f32 = 256 x 16B chunks (2 per thread)
  //   hj tile  = 16*128 f32 = 512 x 16B chunks (4 per thread)
  // => 6 async b128 copies per thread per j-tile.
#define ISSUE_TILE(buf, jb)                                                   \
  {                                                                           \
    const float* ag = afb + (size_t)(jb) * 64;                                \
    const float* hg = hjp + (size_t)(jb) * 128;                               \
    _Pragma("unroll")                                                         \
    for (int c = 0; c < 2; ++c)                                               \
      async_cp16(&saf[buf][(tid + c * 128) * 4], ag + (tid + c * 128) * 4);   \
    _Pragma("unroll")                                                         \
    for (int c = 0; c < 4; ++c)                                               \
      async_cp16(&shj[buf][(tid + c * 128) * 4], hg + (tid + c * 128) * 4);   \
  }

  ISSUE_TILE(0, 0);                       // prologue: tile 0 in flight
  for (int it = 0; it < 16; ++it) {
    const int cur = it & 1;
    const int jb  = it << 4;
    if (it + 1 < 16) {
      ISSUE_TILE(cur ^ 1, jb + 16);       // prefetch next tile (DMA)
      WAIT_ASYNC(6);                      // wait for tile `it` only
    } else {
      WAIT_ASYNC(0);
    }
    __syncthreads();                      // all waves' chunks visible

    const float* ap = &saf[cur][(lane & 15) * 64 + ((lane >> 4) << 4)];
    v16h af0 = load_fragA_f32(ap);        // ds_load + cvt
    v16h af1 = load_fragA_f32(ap + 32);

#pragma unroll
    for (int t = 0; t < 2; ++t) {
      v8f e = {};
      e = wmma_f16(af0, bf[t][0], e);
      e = wmma_f16(af1, bf[t][1], e);
#pragma unroll
      for (int v = 0; v < 8; ++v) {
        const int jl = v + half8;
        const float aj = sadj[jb + jl];
        const float h  = e[v] + shj[cur][jl * 128 + kcol[t]] + cst[t];
        s1[t] += aj * h;
        s2[t] += aj * h * h;
        if (aj > 0.f) {
          mx[t] = fmaxf(mx[t], h);
          mn[t] = fminf(mn[t], h);
        }
      }
    }
    __syncthreads();                      // reads done before buffer reuse
  }
#undef ISSUE_TILE

  // combine the two lane-halves (same column, rows 0-7 vs 8-15)
#pragma unroll
  for (int t = 0; t < 2; ++t) {
    s1[t] += __shfl_xor(s1[t], 16, 32);
    s2[t] += __shfl_xor(s2[t], 16, 32);
    mx[t] = fmaxf(mx[t], __shfl_xor(mx[t], 16, 32));
    mn[t] = fminf(mn[t], __shfl_xor(mn[t], 16, 32));
  }

  const float degc = fmaxf(deg, EPSV);
  const bool  has  = deg > 0.f;
  const float slog = logf(deg + 1.f) / AVGD;
  const float sinv = 1.f / (slog + EPSV);
  float* Xr = X + (size_t)row * 1664;
  Xr[tid] = input[(size_t)row * 128 + tid];   // m_cat[0:128] = input row

  if (lane < 16) {
#pragma unroll
    for (int t = 0; t < 2; ++t) {
      const float mean = s1[t] / degc;
      const float var  = fmaxf(s2[t] / degc - mean * mean, 0.f);
      const float sd   = sqrtf(var + EPSV);
      const float m4[4] = {mean, has ? mx[t] : 0.f, has ? mn[t] : 0.f, sd};
      const int k = kcol[t];
#pragma unroll
      for (int q = 0; q < 4; ++q) {
        Xr[128 +        q * 128 + k] = m4[q];
        Xr[128 +  512 + q * 128 + k] = m4[q] * slog;
        Xr[128 + 1024 + q * 128 + k] = m4[q] * sinv;
      }
    }
  }
}

// ---- GRU elementwise ------------------------------------------------------
__global__ void gru_kernel(const float* __restrict__ gi,
                           const float* __restrict__ gh,
                           const float* __restrict__ hidden,
                           float* __restrict__ out, int total) {
  int t = blockIdx.x * 256 + threadIdx.x;
  if (t >= total) return;
  int row = t >> 7, k = t & 127;
  const float* gir = gi + (size_t)row * 384;
  const float* ghr = gh + (size_t)row * 384;
  float r = 1.f / (1.f + expf(-(gir[k] + ghr[k])));
  float z = 1.f / (1.f + expf(-(gir[128 + k] + ghr[128 + k])));
  float n = tanhf(gir[256 + k] + r * ghr[256 + k]);
  float h = hidden[t];
  out[t] = (1.f - z) * n + z * h;
}

// ---------------------------------------------------------------------------
extern "C" void kernel_launch(void* const* d_in, const int* in_sizes, int n_in,
                              void* d_out, int out_size, void* d_ws, size_t ws_size,
                              hipStream_t stream) {
  (void)in_sizes; (void)n_in; (void)out_size; (void)ws_size;
  const float* input  = (const float*)d_in[0];
  const float* adj    = (const float*)d_in[1];
  const float* adjf   = (const float*)d_in[2];
  const float* hidden = (const float*)d_in[3];
  const float* pre_W  = (const float*)d_in[4];
  const float* pre_b  = (const float*)d_in[5];
  const float* post_W = (const float*)d_in[6];
  const float* post_b = (const float*)d_in[7];
  const float* mix_W  = (const float*)d_in[8];
  const float* mix_b  = (const float*)d_in[9];
  const float* g_Wih  = (const float*)d_in[10];
  const float* g_Whh  = (const float*)d_in[11];

  char* ws = (char*)d_ws;
  // 256B-aligned scratch layout (~24.9 MB total)
  _Float16* WiT  = (_Float16*)(ws + 0);          //  32768 B
  _Float16* WjT  = (_Float16*)(ws + 32768);      //  32768 B
  _Float16* WeT  = (_Float16*)(ws + 65536);      //  16384 B
  _Float16* PT   = (_Float16*)(ws + 81920);      // 425984 B
  _Float16* MT   = (_Float16*)(ws + 507904);     //  32768 B
  _Float16* WihB = (_Float16*)(ws + 540672);     //  98304 B
  _Float16* WhhB = (_Float16*)(ws + 638976);     //  98304 B
  float* hi = (float*)(ws + 737280);             // 2048*128*4
  float* hj = (float*)(ws + 1785856);
  float* X  = (float*)(ws + 2834432);            // 2048*1664*4
  float* Y1 = (float*)(ws + 16465920);
  float* Y2 = (float*)(ws + 17514496);
  float* gi = (float*)(ws + 18563072);           // 2048*384*4
  float* gh = (float*)(ws + 21708800);

  // weight prep (transpose + f16 cast)
  prep_transpose_f16<<<(128 * 128 + 255) / 256, 256, 0, stream>>>(WiT, pre_W, 128, 128);
  prep_transpose_f16<<<(128 * 128 + 255) / 256, 256, 0, stream>>>(WjT, pre_W + 128 * 128, 128, 128);
  prep_transpose_f16<<<(64 * 128 + 255) / 256, 256, 0, stream>>>(WeT, pre_W + 256 * 128, 64, 128);
  prep_transpose_f16<<<(1664 * 128 + 255) / 256, 256, 0, stream>>>(PT, post_W, 1664, 128);
  prep_transpose_f16<<<(128 * 128 + 255) / 256, 256, 0, stream>>>(MT, mix_W, 128, 128);
  prep_cast_f16<<<(384 * 128 + 255) / 256, 256, 0, stream>>>(WihB, g_Wih, 384 * 128);
  prep_cast_f16<<<(384 * 128 + 255) / 256, 256, 0, stream>>>(WhhB, g_Whh, 384 * 128);

  // hi = input @ Wi, hj = input @ Wj
  gemm_wmma<<<256, 128, 0, stream>>>(input, WiT, nullptr, hi, 2048, 128, 128, 0);
  gemm_wmma<<<256, 128, 0, stream>>>(input, WjT, nullptr, hj, 2048, 128, 128, 0);

  // fused PNA aggregation -> X (m_cat, 2048 x 1664)
  pna_kernel<<<2048, 128, 0, stream>>>(input, adj, adjf, hi, hj, pre_b, WeT, X);

  // y = leaky( (X @ post_W + post_b) @ mix_W + mix_b )
  gemm_wmma<<<256, 128, 0, stream>>>(X, PT, post_b, Y1, 2048, 128, 1664, 0);
  gemm_wmma<<<256, 128, 0, stream>>>(Y1, MT, mix_b, Y2, 2048, 128, 128, 1);

  // GRU gates
  gemm_wmma<<<768, 128, 0, stream>>>(Y2, WihB, nullptr, gi, 2048, 384, 128, 0);
  gemm_wmma<<<768, 128, 0, stream>>>(hidden, WhhB, nullptr, gh, 2048, 384, 128, 0);
  gru_kernel<<<(2048 * 128 + 255) / 256, 256, 0, stream>>>(gi, gh, hidden,
                                                           (float*)d_out, 2048 * 128);
}
